// BinaryLSTMCell_395136991615
// MI455X (gfx1250) — matrix operand order repaired and checked by reading, not compiled, not run on hardware
//
#include <hip/hip_runtime.h>
#include <hip/hip_bf16.h>

// Problem constants (from reference)
#define BB 32
#define SS 512
#define FF 512
#define HH 1024
#define NG (4 * HH)   // 4096 combined gate columns, gate order: i, f, o, g
#define NBLK 64       // persistent-kernel blocks (512 waves = 512 C tiles)

// TDM-staged LDS tile: 32 KB payload + 16 B pad per 1 KB interval = 33280 B/wave
#define LDS_WAVE_BYTES 33280u

typedef __attribute__((ext_vector_type(16))) _Float16 v16h;
typedef __attribute__((ext_vector_type(8)))  _Float16 v8h;
typedef __attribute__((ext_vector_type(8)))  float    v8f;
typedef __attribute__((ext_vector_type(4)))  unsigned int v4u;
typedef __attribute__((ext_vector_type(4)))  int      v4i;
typedef __attribute__((ext_vector_type(8)))  int      v8i;

// ---------------------------------------------------------------------------
// alpha = mean(|w|), one block per call
// ---------------------------------------------------------------------------
__global__ void k_absmean(const float* __restrict__ p, int n, float* __restrict__ out) {
    __shared__ float red[256];
    float s = 0.f;
    for (int i = threadIdx.x; i < n; i += blockDim.x) s += fabsf(p[i]);
    red[threadIdx.x] = s;
    __syncthreads();
    for (int w = 128; w > 0; w >>= 1) {
        if ((int)threadIdx.x < w) red[threadIdx.x] += red[threadIdx.x + w];
        __syncthreads();
    }
    if (threadIdx.x == 0) *out = red[0] / (float)n;
}

// ---------------------------------------------------------------------------
// dst = fp16(copysign(alpha, w))  — binarized weight, exactly representable
// ---------------------------------------------------------------------------
__global__ void k_binarize(const float* __restrict__ w, _Float16* __restrict__ dst,
                           const float* __restrict__ alpha, int n) {
    int i = blockIdx.x * blockDim.x + threadIdx.x;
    if (i >= n) return;
    float a = *alpha;
    dst[i] = (_Float16)copysignf(a, w[i]);
}

// ---------------------------------------------------------------------------
// One-time fp16 copy of the activations (removes per-tile f32->f16 converts
// from the GEMM inner loop; A-bytes halve and each byte is converted once,
// not 256 times).
// ---------------------------------------------------------------------------
__global__ void k_cvtx(const float* __restrict__ X, _Float16* __restrict__ X16, int n) {
    int i = blockIdx.x * blockDim.x + threadIdx.x;
    if (i >= n) return;
    X16[i] = (_Float16)X[i];
}

// ---------------------------------------------------------------------------
// biasv[q*H+h] = bin(b_q_input)[h] + bin(b_q_hidden)[h]
// ---------------------------------------------------------------------------
__global__ void k_biasvec(const float* __restrict__ bii, const float* __restrict__ bhi,
                          const float* __restrict__ bif, const float* __restrict__ bhf,
                          const float* __restrict__ bio, const float* __restrict__ bho,
                          const float* __restrict__ big, const float* __restrict__ bhg,
                          const float* __restrict__ alpha, float* __restrict__ biasv) {
    int i = blockIdx.x * blockDim.x + threadIdx.x;
    if (i >= NG) return;
    int q = i >> 10, hh = i & (HH - 1);
    const float *p1, *p2; float a1, a2;
    switch (q) {
        case 0:  p1 = bii; p2 = bhi; a1 = alpha[2];  a2 = alpha[3];  break;
        case 1:  p1 = bif; p2 = bhf; a1 = alpha[6];  a2 = alpha[7];  break;
        case 2:  p1 = bio; p2 = bho; a1 = alpha[10]; a2 = alpha[11]; break;
        default: p1 = big; p2 = bhg; a1 = alpha[14]; a2 = alpha[15]; break;
    }
    biasv[i] = copysignf(a1, p1[hh]) + copysignf(a2, p2[hh]);
}

// ---------------------------------------------------------------------------
// Initialize recurrent state + zero the grid-barrier counters
// ---------------------------------------------------------------------------
__global__ void k_init_state(const float* __restrict__ h0, const float* __restrict__ c0,
                             _Float16* __restrict__ h16, float* __restrict__ c,
                             int* __restrict__ bar, int n) {
    int i = blockIdx.x * blockDim.x + threadIdx.x;
    if (i >= n) return;
    h16[i] = (_Float16)h0[i];
    c[i] = c0[i];
    if (i < 2 * SS) bar[i] = 0;
}

// ---------------------------------------------------------------------------
// Phase 1: XPRE[s][b][n] = x[b][s][:] @ WX[n][:]  for all 4 gates at once.
// One wave computes one 16x16 C tile with V_WMMA_F32_16X16X32_F16.
// M = B*S = 16384 (m = b*S+s), N = 4096, K = F = 512.
// ---------------------------------------------------------------------------
__global__ void __launch_bounds__(256)
k_xproj(const _Float16* __restrict__ X16,   // (B,S,F) fp16
        const _Float16* __restrict__ WX,    // (4,H,F) binarized fp16
        _Float16* __restrict__ XPRE) {      // (S,B,NG) fp16
    const int wave = (blockIdx.x << 3) | (threadIdx.x >> 5);  // 8 waves/block
    const int lane = threadIdx.x & 31;
    const int mt = wave >> 8;        // 0..1023  (M tiles)
    const int nt = wave & 255;       // 0..255   (N tiles)
    const int hi = lane >> 4;        // lane half-group
    const int l4 = lane & 15;

    const int m = (mt << 4) + l4;                  // m = b*S + s (flat row)
    const _Float16* __restrict__ arow = X16 + (size_t)m * FF;

    const int n = (nt << 4) + l4;
    const int q = n >> 10, hr = n & (HH - 1);
    const _Float16* __restrict__ brow = WX + ((size_t)q * HH + hr) * FF;

    v8f acc = {};
    for (int kb = 0; kb < FF; kb += 32) {
        v8h a0 = *(const v8h*)(arow + kb + hi * 16);
        v8h a1 = *(const v8h*)(arow + kb + hi * 16 + 8);
        v16h a = __builtin_shufflevector(a0, a1, 0, 1, 2, 3, 4, 5, 6, 7,
                                                 8, 9, 10, 11, 12, 13, 14, 15);
        v16h b = *(const v16h*)(brow + kb + hi * 16);
        acc = __builtin_amdgcn_wmma_f32_16x16x32_f16(false, a, false, b,
                                                     (short)0, acc, false, false);
    }

#pragma unroll
    for (int r = 0; r < 8; ++r) {
        const int m2 = (mt << 4) + r + hi * 8;
        const int b2 = m2 >> 9;          // / S
        const int s2 = m2 & (SS - 1);    // % S
        XPRE[(((size_t)s2 * BB + b2) * NG) + n] = (_Float16)acc[r];
    }
}

// ---------------------------------------------------------------------------
// Grid-wide barrier: one single-use atomic counter per barrier instance.
// Spin waves sleep between polls (s_sleep) to free SALU/L2 bandwidth.
// ---------------------------------------------------------------------------
__device__ __forceinline__ void grid_barrier(int* __restrict__ bar, int idx) {
    __threadfence();            // make this thread's global stores visible
    __syncthreads();            // order all threads' fences before the signal
    if (threadIdx.x == 0) {
        atomicAdd(&bar[idx], 1);
        while (__hip_atomic_load(&bar[idx], __ATOMIC_RELAXED,
                                 __HIP_MEMORY_SCOPE_AGENT) < NBLK) {
            __builtin_amdgcn_s_sleep(1);
        }
    }
    __syncthreads();
    __threadfence();            // acquire side
}

// ---------------------------------------------------------------------------
// Phase 2: persistent recurrent kernel. 64 blocks x 256 threads = 512 waves,
// one wave per 16x16 tile of the (32 x 4096) gate pre-activation matrix.
//
// Each wave's B tile is 16 CONTIGUOUS Wh rows (32 KB) — staged into LDS once
// by the Tensor Data Mover (tensor_load_to_lds, TENSORcnt), with TDM LDS
// padding (16 B per 1 KB) so the padded row stride (520 DWORDs, = 8 mod 64
// banks) avoids the 2048-byte-stride bank-0 pileup. All 512 timesteps then
// read B from LDS.  Per-step: WMMA GEMM -> grid barrier -> LSTM pointwise ->
// grid barrier.
// ---------------------------------------------------------------------------
__global__ void __launch_bounds__(256)
k_recur(_Float16* __restrict__ h16,           // (B,H) fp16, updated per step
        const _Float16* __restrict__ WH,      // (4,H,H) binarized fp16
        const _Float16* __restrict__ XPRE,    // (S,B,NG) fp16
        const float* __restrict__ biasv,      // (NG)
        float* __restrict__ gates,            // (B,NG) f32 scratch
        float* __restrict__ cstate,           // (B,H) f32
        float* __restrict__ hidden,           // (B,S,H) = d_out
        int* __restrict__ bar) {              // 2*S counters, pre-zeroed
    extern __shared__ _Float16 Bsm[];         // 8 waves * 33280 B = 260 KB LDS

    const int wave = threadIdx.x >> 5;
    const int lane = threadIdx.x & 31;
    const int tile = (blockIdx.x << 3) | wave;  // 0..511
    const int mt = tile >> 8;                   // 0..1
    const int nt = tile & 255;                  // 0..255
    const int hi = lane >> 4;
    const int l4 = lane & 15;

    // ---- TDM: DMA this wave's 16 contiguous Wh rows (32 KB) into LDS ----
    {
        const int n0 = nt << 4;
        const int q0 = n0 >> 10, hr0 = n0 & (HH - 1);
        const unsigned long long gaddr =
            (unsigned long long)(const void*)(WH + ((size_t)q0 * HH + hr0) * HH);
        const unsigned ldsa = (unsigned)__builtin_amdgcn_groupstaticsize()
                            + (unsigned)wave * LDS_WAVE_BYTES;

        // D# group 0: count=1 | lds_addr | global_addr[56:0] | type=2
        v4u g0 = { 1u, ldsa, (unsigned)gaddr,
                   (unsigned)((gaddr >> 32) & 0x01FFFFFFu) | (2u << 30) };
        // D# group 1: data_size=3 (8B), pad_enable, pad_interval=7 (256 dw),
        // pad_amount=3 (4 dw); tensor_dim0=tile_dim0=4096 units; dims1=1;
        // tensor_dim0_stride=4096.
        v8i g1 = { (int)((3u << 16) | (1u << 20) | (7u << 22) | (3u << 25)),
                   0x10000000, 0x00010000, 0x10000000, 1, 0x1000, 0, 0 };
        v4i gz = { 0, 0, 0, 0 };
#if defined(__clang_major__) && (__clang_major__ >= 23)
        v8i gz8 = { 0, 0, 0, 0, 0, 0, 0, 0 };
        __builtin_amdgcn_tensor_load_to_lds(g0, g1, gz, gz, gz8, 0);
#else
        __builtin_amdgcn_tensor_load_to_lds(g0, g1, gz, gz, 0);
#endif
        __builtin_amdgcn_s_wait_tensorcnt(0);
    }

    const char* __restrict__ myB = (const char*)Bsm + (size_t)wave * LDS_WAVE_BYTES;

    const int n = (nt << 4) + l4;
    const int m = (mt << 4) + l4;               // batch row for A
    const _Float16* __restrict__ arow = h16 + (size_t)m * HH;
    const float bn = biasv[n];
    const int tid = blockIdx.x * 256 + threadIdx.x;   // 0..16383 (pointwise)

    for (int t = 0; t < SS; ++t) {
        const _Float16* __restrict__ xpre_t = XPRE + (size_t)t * BB * NG;

        // Prefetch next step's 256 KB XPRE slice behind this step's compute.
        if (t + 1 < SS) {
            const _Float16* nx = XPRE + (size_t)(t + 1) * BB * NG;
            __builtin_prefetch(nx + (size_t)tid * 8, 0, 1);
        }

        // Accumulator initialized with precomputed input projection + bias
        v8f acc;
#pragma unroll
        for (int r = 0; r < 8; ++r) {
            const int row = (mt << 4) + r + hi * 8;
            acc[r] = (float)xpre_t[(size_t)row * NG + n] + bn;
        }

        // K loop: A from global h16 (hot in WGP$), B from TDM-padded LDS.
        for (int kbi = 0; kbi < 32; ++kbi) {
            const int kb = kbi * 32;
            v8h a0 = *(const v8h*)(arow + kb + hi * 16);
            v8h a1 = *(const v8h*)(arow + kb + hi * 16 + 8);
            v16h a = __builtin_shufflevector(a0, a1, 0, 1, 2, 3, 4, 5, 6, 7,
                                                     8, 9, 10, 11, 12, 13, 14, 15);
            // Unpadded byte offset within the wave's row-major tile; add
            // 16 B per completed 1 KB interval (TDM pad).  Fragments are
            // 32 B-aligned pre-pad, so a pad never splits one.
            const unsigned Xo = (unsigned)(l4 * 2048 + kbi * 64 + hi * 32);
            const char* bp = myB + Xo + ((Xo >> 10) << 4);
            v8h b0 = *(const v8h*)(bp);
            v8h b1 = *(const v8h*)(bp + 16);
            v16h b = __builtin_shufflevector(b0, b1, 0, 1, 2, 3, 4, 5, 6, 7,
                                                     8, 9, 10, 11, 12, 13, 14, 15);
            acc = __builtin_amdgcn_wmma_f32_16x16x32_f16(false, a, false, b,
                                                         (short)0, acc, false, false);
        }

#pragma unroll
        for (int r = 0; r < 8; ++r) {
            const int row = (mt << 4) + r + hi * 8;
            gates[(size_t)row * NG + n] = acc[r];
        }

        grid_barrier(bar, 2 * t);       // all gate preacts visible

        // LSTM pointwise update: 2 of 32768 elements per thread.
#pragma unroll
        for (int e = 0; e < 2; ++e) {
            const int i = tid + e * 16384;
            const int b = i >> 10, hh = i & (HH - 1);
            const float* g = gates + (size_t)b * NG;
            const float ip = g[hh];
            const float fp = g[HH + hh];
            const float op = g[2 * HH + hh];
            const float gp = g[3 * HH + hh];
            const float si = 1.0f / (1.0f + __expf(-ip));
            const float sf = 1.0f / (1.0f + __expf(-fp));
            const float so = 1.0f / (1.0f + __expf(-op));
            const float c2 = sf * cstate[i] + si * tanhf(gp);
            const float h2 = so * tanhf(c2);
            cstate[i] = c2;
            h16[i] = (_Float16)h2;
            hidden[((size_t)b * SS + t) * HH + hh] = h2;
        }

        grid_barrier(bar, 2 * t + 1);   // h16/cstate updated for next step
    }
}

// ---------------------------------------------------------------------------
// Final: append h (= hidden[:, S-1, :]) and c to the output tail
// ---------------------------------------------------------------------------
__global__ void k_final(const float* __restrict__ hidden, const float* __restrict__ cstate,
                        float* __restrict__ hout, float* __restrict__ cout) {
    int i = blockIdx.x * blockDim.x + threadIdx.x;
    if (i >= BB * HH) return;
    const int b = i >> 10, hh = i & (HH - 1);
    hout[i] = hidden[((size_t)b * SS + (SS - 1)) * HH + hh];
    cout[i] = cstate[i];
}

// ---------------------------------------------------------------------------
// Host-side launch sequence
// ---------------------------------------------------------------------------
extern "C" void kernel_launch(void* const* d_in, const int* in_sizes, int n_in,
                              void* d_out, int out_size, void* d_ws, size_t ws_size,
                              hipStream_t stream) {
    (void)in_sizes; (void)n_in; (void)out_size; (void)ws_size;

    const float* X  = (const float*)d_in[0];
    const float* h0 = (const float*)d_in[1];
    const float* c0 = (const float*)d_in[2];
    // inputs 3..18: [w_ii,w_hi,b_ii,b_hi, w_if,w_hf,b_if,b_hf,
    //                w_io,w_ho,b_io,b_ho, w_ig,w_hg,b_ig,b_hg]

    char* ws = (char*)d_ws;
    const size_t ALPHA_OFF = 0;                                    // 16 f32 (pad 256)
    const size_t WX_OFF    = 256;                                  // 4*H*F fp16 = 4 MB
    const size_t WH_OFF    = WX_OFF   + (size_t)4 * HH * FF * 2;   // 4*H*H fp16 = 8 MB
    const size_t BIAS_OFF  = WH_OFF   + (size_t)4 * HH * HH * 2;   // NG f32
    const size_t H16_OFF   = BIAS_OFF + (size_t)NG * 4;            // B*H fp16
    const size_t C_OFF     = H16_OFF  + (size_t)BB * HH * 2;       // B*H f32
    const size_t GATES_OFF = C_OFF    + (size_t)BB * HH * 4;       // B*NG f32
    const size_t BAR_OFF   = GATES_OFF + (size_t)BB * NG * 4;      // 2*S ints
    const size_t X16_OFF   = BAR_OFF  + (size_t)2 * SS * 4;        // B*S*F fp16 = 16 MB
    const size_t XPRE_OFF  = X16_OFF  + (size_t)BB * SS * FF * 2;  // S*B*NG fp16 = 134 MB

    float*     alpha = (float*)(ws + ALPHA_OFF);
    _Float16*  WX    = (_Float16*)(ws + WX_OFF);
    _Float16*  WH    = (_Float16*)(ws + WH_OFF);
    float*     biasv = (float*)(ws + BIAS_OFF);
    _Float16*  h16   = (_Float16*)(ws + H16_OFF);
    float*     cst   = (float*)(ws + C_OFF);
    float*     gates = (float*)(ws + GATES_OFF);
    int*       bar   = (int*)(ws + BAR_OFF);
    _Float16*  X16   = (_Float16*)(ws + X16_OFF);
    _Float16*  XPRE  = (_Float16*)(ws + XPRE_OFF);

    float* out   = (float*)d_out;                 // hidden_seq (B,S,H)
    float* h_out = out + (size_t)BB * SS * HH;
    float* c_out = h_out + (size_t)BB * HH;

    // 1) alphas for all 16 weight/bias tensors (alpha[k] for input index 3+k)
    for (int k = 0; k < 16; ++k) {
        const float* p = (const float*)d_in[3 + k];
        int n;
        switch (k & 3) { case 0: n = HH * FF; break;   // w_i* (H,F)
                         case 1: n = HH * HH; break;   // w_h* (H,H)
                         default: n = HH; break; }     // biases (H,1)
        k_absmean<<<1, 256, 0, stream>>>(p, n, alpha + k);
    }

    // 2) binarize weights into fp16 (gate order i,f,o,g matches input order)
    for (int q = 0; q < 4; ++q) {
        const float* wi = (const float*)d_in[3 + 4 * q];
        const float* wh = (const float*)d_in[4 + 4 * q];
        k_binarize<<<(HH * FF + 255) / 256, 256, 0, stream>>>(
            wi, WX + (size_t)q * HH * FF, alpha + 4 * q, HH * FF);
        k_binarize<<<(HH * HH + 255) / 256, 256, 0, stream>>>(
            wh, WH + (size_t)q * HH * HH, alpha + 4 * q + 1, HH * HH);
    }

    // 3) activations to fp16 (one-time) + combined bias vector
    k_cvtx<<<(BB * SS * FF + 255) / 256, 256, 0, stream>>>(X, X16, BB * SS * FF);
    k_biasvec<<<(NG + 255) / 256, 256, 0, stream>>>(
        (const float*)d_in[5],  (const float*)d_in[6],
        (const float*)d_in[9],  (const float*)d_in[10],
        (const float*)d_in[13], (const float*)d_in[14],
        (const float*)d_in[17], (const float*)d_in[18],
        alpha, biasv);

    // 4) state init + barrier-counter reset (every call -> graph-replay safe)
    k_init_state<<<(BB * HH + 255) / 256, 256, 0, stream>>>(h0, c0, h16, cst, bar, BB * HH);

    // 5) all input projections in one big WMMA GEMM (fully parallel)
    {
        const int tiles = (BB * SS / 16) * (NG / 16);   // 1024 * 256
        k_xproj<<<tiles / 8, 256, 0, stream>>>(X16, WX, XPRE);
    }

    // 6) persistent recurrent kernel: all 512 timesteps in ONE launch.
    //    TDM stages each wave's Wh tile into (padded) dynamic LDS.
    k_recur<<<NBLK, 256, 8 * LDS_WAVE_BYTES, stream>>>(
        h16, WH, XPRE, biasv, gates, cst, out, bar);

    // 7) final h, c
    k_final<<<(BB * HH + 255) / 256, 256, 0, stream>>>(out, cst, h_out, c_out);
}
